// LocalGeoAgg_43800076485392
// MI455X (gfx1250) — matrix operand order repaired and checked by reading, not compiled, not run on hardware
//
#include <hip/hip_runtime.h>
#include <hip/hip_bf16.h>
#include <math.h>

// ---------------------------------------------------------------------------
// LocalGeoAgg for MI455X (gfx1250, wave32, WMMA).
// GEMMs run as v_wmma_f32_16x16x32_bf16 (fp32 in -> bf16 operands, f32 acc).
// BatchNorm statistics are global over all B*G*K points, so the pipeline is a
// chain of fused (apply-prev-BN -> GEMM -> accumulate-stats) kernels with tiny
// finalize kernels in between. Intermediates live in bf16 workspace to halve
// HBM traffic (23.3 TB/s roofline: ~1 GB moved total).
// ---------------------------------------------------------------------------

#define BSZ   8
#define GSZ   2048
#define KSZ   32
#define NPTS  (BSZ * GSZ * KSZ)   // 524288 points
#define INCH  131                 // 67 knn_feat + 64 lc_feat
#define CKPAD 160                 // pad to 5 * 32 for WMMA K-steps
#define OUTD  128
#define MIDD  64

typedef __attribute__((ext_vector_type(16))) __bf16 v16bf;
typedef __attribute__((ext_vector_type(8)))  __bf16 bf16x8;
typedef __attribute__((ext_vector_type(8)))  float  v8f;

__device__ __forceinline__ float  bf2f(__bf16 v) { return (float)v; }
__device__ __forceinline__ __bf16 f2bf(float v)  { return (__bf16)v; }

// A-matrix fragment (16x32 bf16, row-major source, stride lda).
// ISA 7.12.2: lane(hi,lo) row M=lo holds K in [8*hi,8*hi+8) and [16+8*hi,+8).
__device__ __forceinline__ v16bf load_a(const __bf16* A, int lda) {
  const int lane = threadIdx.x & 31;
  const int lo = lane & 15, hi = lane >> 4;
  const __bf16* r = A + (size_t)lo * lda + hi * 8;
  v16bf f;
#pragma unroll
  for (int j = 0; j < 8; ++j) { f[j] = r[j]; f[8 + j] = r[16 + j]; }
  return f;
}

// B-matrix fragment (32x16 bf16) from column-major storage (column n = one
// point's feature vector, contiguous over k). lane(hi,lo): col N=lo, K in
// [8*hi,8*hi+8) and [16+8*hi,+8) (matches A fragment).
__device__ __forceinline__ v16bf load_b_col(const __bf16* Bc, int ldb) {
  const int lane = threadIdx.x & 31;
  const int lo = lane & 15, hi = lane >> 4;
  const __bf16* c = Bc + (size_t)lo * ldb + hi * 8;
  v16bf f;
#pragma unroll
  for (int j = 0; j < 8; ++j) { f[j] = c[j]; f[8 + j] = c[16 + j]; }
  return f;
}

__device__ __forceinline__ v8f wmma_bf16(v16bf a, v16bf b, v8f c) {
  // D = A x B + C, 16x16x32, f32 accumulate
  return __builtin_amdgcn_wmma_f32_16x16x32_bf16(false, a, false, b,
                                                 (short)0, c, false, false);
}

// D layout: lane(hi,lo) -> point n = lo, channels m = 8*hi + r (r = VGPR idx).
// Per-channel sum/sumsq: accumulate per lane, butterfly over the 16 "lo"
// lanes (width-16 shuffles keep the two hi-halves separate), then atomicAdd.
__device__ __forceinline__ void stat_commit(float s[8], float q[8],
                                            float* stats, int cbase, int lo) {
#pragma unroll
  for (int r = 0; r < 8; ++r) {
#pragma unroll
    for (int m = 1; m < 16; m <<= 1) {
      s[r] += __shfl_xor(s[r], m, 16);
      q[r] += __shfl_xor(q[r], m, 16);
    }
  }
  if (lo == 0) {
#pragma unroll
    for (int r = 0; r < 8; ++r) {
      atomicAdd(&stats[cbase + r], s[r]);
      atomicAdd(&stats[128 + cbase + r], q[r]);
    }
  }
}

// --------------------------- scalar std of rel0 ----------------------------
__global__ void __launch_bounds__(256) k_relstats(const float* __restrict__ knn_xyz,
                                                  const float* __restrict__ lc_xyz,
                                                  float* __restrict__ hdr) {
  __shared__ float ss[256], sq[256];
  float s = 0.f, q = 0.f;
  const int N = NPTS * 3;
  for (int i = blockIdx.x * 256 + threadIdx.x; i < N; i += gridDim.x * 256) {
    int p = i / 3, d = i - p * 3;
    int bg = p >> 5;
    float v = knn_xyz[i] - lc_xyz[bg * 3 + d];
    s += v; q += v * v;
  }
  ss[threadIdx.x] = s; sq[threadIdx.x] = q;
  __syncthreads();
  for (int off = 128; off; off >>= 1) {
    if (threadIdx.x < off) {
      ss[threadIdx.x] += ss[threadIdx.x + off];
      sq[threadIdx.x] += sq[threadIdx.x + off];
    }
    __syncthreads();
  }
  if (threadIdx.x == 0) { atomicAdd(&hdr[0], ss[0]); atomicAdd(&hdr[1], sq[0]); }
}

__global__ void k_std(float* hdr) {
  if (threadIdx.x == 0) {
    double n = (double)NPTS * 3.0;
    double s = hdr[0], q = hdr[1];
    double var = (q - s * s / n) / (n - 1.0);   // ddof = 1
    hdr[2] = (float)(sqrt(var) + 1e-5);
  }
}

// --------------------------- gaussian point weight -------------------------
__global__ void __launch_bounds__(256) k_gaussw(const float* __restrict__ knn_xyz,
                                                const float* __restrict__ lc_xyz,
                                                const float* __restrict__ hdr,
                                                float* __restrict__ gw) {
  int p = blockIdx.x * 256 + threadIdx.x;
  if (p >= NPTS) return;
  int bg = p >> 5;
  float inv_std = 1.f / hdr[2];
  float acc = 0.f;
#pragma unroll
  for (int d = 0; d < 3; ++d) {
    float lc = lc_xyz[bg * 3 + d];
    float rn = (knn_xyz[(size_t)p * 3 + d] - lc) * inv_std - lc;
    acc += rn * rn;
  }
  gw[p] = expf(-sqrtf(acc) * 0.5f);   // 2*SIGMA^2 = 2
}

// ------------------- weight pre-conversion fp32 -> bf16 --------------------
__global__ void __launch_bounds__(256) k_prep(const float* __restrict__ w1,
                                              const float* __restrict__ wd,
                                              const float* __restrict__ wu,
                                              __bf16* __restrict__ w1bf,
                                              __bf16* __restrict__ wdbf,
                                              __bf16* __restrict__ wubf) {
  int i = blockIdx.x * 256 + threadIdx.x;
  if (i < OUTD * CKPAD) {                       // w1 padded 131 -> 160
    int o = i / CKPAD, c = i - o * CKPAD;
    w1bf[i] = f2bf(c < INCH ? w1[o * INCH + c] : 0.f);
  }
  int j = i - OUTD * CKPAD;
  if (j >= 0 && j < 2 * MIDD * OUTD) wdbf[j] = f2bf(wd[j]);
  int k = j - 2 * MIDD * OUTD;
  if (k >= 0 && k < 2 * OUTD * MIDD) wubf[k] = f2bf(wu[k]);
}

// ---- GEMM1: x_pre = w1 * expanded(67 knn_feat ++ 64 lc_feat), + bn1 stats --
__global__ void __launch_bounds__(256) k_gemm1(const float* __restrict__ knn_feat,
                                               const float* __restrict__ lc_feat,
                                               const __bf16* __restrict__ w1bf,
                                               __bf16* __restrict__ xpre,
                                               float* __restrict__ stats) {
  __shared__ __align__(32) __bf16 sB[64 * CKPAD];   // 64 points x 160 ch
  const int tid = threadIdx.x;
  const int p0 = blockIdx.x * 64;
  // stage expanded features: 64*20 runs of 8 channels, 5 per thread
#pragma unroll
  for (int it = 0; it < 5; ++it) {
    int run = tid + it * 256;
    int pl = run / 20;
    int c0 = (run - pl * 20) * 8;
    int p = p0 + pl;
    int bg = p >> 5;
#pragma unroll
    for (int j = 0; j < 8; ++j) {
      int c = c0 + j;
      float v = 0.f;
      if (c < 67)        v = knn_feat[(size_t)p * 67 + c];
      else if (c < INCH) v = lc_feat[(size_t)bg * 64 + (c - 67)];
      sB[pl * CKPAD + c] = f2bf(v);
    }
  }
  __syncthreads();

  const int wave = tid >> 5, lane = tid & 31;
  const int lo = lane & 15, hi = lane >> 4;
  const int mtile = wave;                       // 8 waves -> 8 channel tiles
  v16bf afr[5];
#pragma unroll
  for (int kt = 0; kt < 5; ++kt)
    afr[kt] = load_a(w1bf + (size_t)(mtile * 16) * CKPAD + kt * 32, CKPAD);

  float s[8] = {0,0,0,0,0,0,0,0}, q[8] = {0,0,0,0,0,0,0,0};
#pragma unroll
  for (int nt = 0; nt < 4; ++nt) {
    v8f acc = {};
#pragma unroll
    for (int kt = 0; kt < 5; ++kt)
      acc = wmma_bf16(afr[kt], load_b_col(&sB[(nt * 16) * CKPAD + kt * 32], CKPAD), acc);
    int p = p0 + nt * 16 + lo;
    int cbase = mtile * 16 + hi * 8;
    bf16x8 out;
#pragma unroll
    for (int r = 0; r < 8; ++r) {
      float v = acc[r];
      out[r] = f2bf(v);
      s[r] += v; q[r] += v * v;
    }
    *(bf16x8*)(xpre + (size_t)p * OUTD + cbase) = out;
  }
  stat_commit(s, q, stats, mtile * 16 + hi * 8, lo);
}

// ---- down stage: x = epilogue(vpre); h_pre = wd*x + bd ; stats(h_pre) -----
// mode A (after bn1):   x = relu(a*v+b) * gw[p]          (gw != null)
// mode B (after up-bn): x = relu(a*v+b + xres)           (xres != null)
__global__ void __launch_bounds__(256) k_down(const __bf16* __restrict__ vpre,
                                              const float* __restrict__ ab,
                                              const float* __restrict__ gw,
                                              const __bf16* xres,
                                              __bf16* xcur,
                                              const __bf16* __restrict__ wdbf,
                                              const float* __restrict__ bd,
                                              __bf16* __restrict__ hpre,
                                              float* __restrict__ stats) {
  __shared__ __align__(32) __bf16 sX[64 * OUTD];
  __shared__ float sA[OUTD], sC[OUTD];
  const int tid = threadIdx.x;
  if (tid < OUTD) { sA[tid] = ab[tid]; sC[tid] = ab[128 + tid]; }
  __syncthreads();

  const int p0 = blockIdx.x * 64;
#pragma unroll
  for (int it = 0; it < 4; ++it) {             // 64*16 runs of 8ch, 4/thread
    int run = tid + it * 256;
    int pl = run >> 4;
    int c0 = (run & 15) * 8;
    int p = p0 + pl;
    bf16x8 vin = *(const bf16x8*)(vpre + (size_t)p * OUTD + c0);
    bf16x8 res{};
    if (xres) res = *(const bf16x8*)(xres + (size_t)p * OUTD + c0);
    float gwv = gw ? gw[p] : 1.f;
    bf16x8 o;
#pragma unroll
    for (int j = 0; j < 8; ++j) {
      int c = c0 + j;
      float v = sA[c] * bf2f(vin[j]) + sC[c];
      if (xres) v += bf2f(res[j]);
      v = v > 0.f ? v : 0.f;
      v *= gwv;
      o[j] = f2bf(v);
      sX[pl * OUTD + c] = o[j];
    }
    *(bf16x8*)(xcur + (size_t)p * OUTD + c0) = o;  // keep for residual
  }
  __syncthreads();

  const int wave = tid >> 5, lane = tid & 31;
  const int lo = lane & 15, hi = lane >> 4;
  const int mtile = wave >> 1;                 // 4 MID tiles, 2 waves each
  const int ntb = (wave & 1) * 2;
  v16bf afr[4];
#pragma unroll
  for (int kt = 0; kt < 4; ++kt)
    afr[kt] = load_a(wdbf + (size_t)(mtile * 16) * OUTD + kt * 32, OUTD);

  float s[8] = {0,0,0,0,0,0,0,0}, q[8] = {0,0,0,0,0,0,0,0};
#pragma unroll
  for (int ni = 0; ni < 2; ++ni) {
    int nt = ntb + ni;
    v8f acc = {};
#pragma unroll
    for (int kt = 0; kt < 4; ++kt)
      acc = wmma_bf16(afr[kt], load_b_col(&sX[(nt * 16) * OUTD + kt * 32], OUTD), acc);
    int p = p0 + nt * 16 + lo;
    int cbase = mtile * 16 + hi * 8;
    bf16x8 out;
#pragma unroll
    for (int r = 0; r < 8; ++r) {
      float v = acc[r] + bd[cbase + r];
      out[r] = f2bf(v);
      s[r] += v; q[r] += v * v;
    }
    *(bf16x8*)(hpre + (size_t)p * MIDD + cbase) = out;
  }
  stat_commit(s, q, stats, mtile * 16 + hi * 8, lo);
}

// ---- up stage: h = relu(a*v+b); u_pre = wu*h + bu ; stats(u_pre) ----------
__global__ void __launch_bounds__(256) k_up(const __bf16* __restrict__ vpre,
                                            const float* __restrict__ ab,
                                            const __bf16* __restrict__ wubf,
                                            const float* __restrict__ bu,
                                            __bf16* __restrict__ upre,
                                            float* __restrict__ stats) {
  __shared__ __align__(32) __bf16 sH[64 * MIDD];
  __shared__ float sA[MIDD], sC[MIDD];
  const int tid = threadIdx.x;
  if (tid < MIDD) { sA[tid] = ab[tid]; sC[tid] = ab[128 + tid]; }
  __syncthreads();

  const int p0 = blockIdx.x * 64;
#pragma unroll
  for (int it = 0; it < 2; ++it) {             // 64*8 runs of 8ch, 2/thread
    int run = tid + it * 256;
    int pl = run >> 3;
    int c0 = (run & 7) * 8;
    int p = p0 + pl;
    bf16x8 vin = *(const bf16x8*)(vpre + (size_t)p * MIDD + c0);
#pragma unroll
    for (int j = 0; j < 8; ++j) {
      int c = c0 + j;
      float v = sA[c] * bf2f(vin[j]) + sC[c];
      sH[pl * MIDD + c] = f2bf(v > 0.f ? v : 0.f);
    }
  }
  __syncthreads();

  const int wave = tid >> 5, lane = tid & 31;
  const int lo = lane & 15, hi = lane >> 4;
  const int mtile = wave;                      // 8 OUT tiles
  v16bf afr[2];
#pragma unroll
  for (int kt = 0; kt < 2; ++kt)
    afr[kt] = load_a(wubf + (size_t)(mtile * 16) * MIDD + kt * 32, MIDD);

  float s[8] = {0,0,0,0,0,0,0,0}, q[8] = {0,0,0,0,0,0,0,0};
#pragma unroll
  for (int nt = 0; nt < 4; ++nt) {
    v8f acc = {};
#pragma unroll
    for (int kt = 0; kt < 2; ++kt)
      acc = wmma_bf16(afr[kt], load_b_col(&sH[(nt * 16) * MIDD + kt * 32], MIDD), acc);
    int p = p0 + nt * 16 + lo;
    int cbase = mtile * 16 + hi * 8;
    bf16x8 out;
#pragma unroll
    for (int r = 0; r < 8; ++r) {
      float v = acc[r] + bu[cbase + r];
      out[r] = f2bf(v);
      s[r] += v; q[r] += v * v;
    }
    *(bf16x8*)(upre + (size_t)p * OUTD + cbase) = out;
  }
  stat_commit(s, q, stats, mtile * 16 + hi * 8, lo);
}

// ------------------------ BN finalize: per-channel affine -------------------
__global__ void k_finalize(const float* __restrict__ stats,
                           const float* __restrict__ gamma,
                           const float* __restrict__ beta,
                           float* __restrict__ ab, int C) {
  int c = threadIdx.x;
  if (c >= C) return;
  const float n = (float)NPTS;
  float mean = stats[c] / n;
  float var = stats[128 + c] / n - mean * mean;   // ddof = 0
  float a = gamma[c] * rsqrtf(var + 1e-5f);
  ab[c] = a;
  ab[128 + c] = beta[c] - mean * a;
}

// -------- final: out[b][c][g][k] = relu(a*u_pre + b + x1), fp32 ------------
__global__ void __launch_bounds__(256) k_final(const __bf16* __restrict__ upre,
                                               const __bf16* __restrict__ xcur,
                                               const float* __restrict__ ab,
                                               float* __restrict__ out) {
  int idx = blockIdx.x * 256 + threadIdx.x;     // 8 * 128 * 2048 * 4 runs
  int kr = idx & 3;
  int g  = (idx >> 2) & 2047;
  int c  = (idx >> 13) & 127;
  int b  = idx >> 20;
  float a = ab[c], bb = ab[128 + c];
  size_t pbase = ((size_t)(b * GSZ + g)) * KSZ + kr * 8;
  size_t obase = (((size_t)(b * OUTD + c)) * GSZ + g) * KSZ + kr * 8;
#pragma unroll
  for (int j = 0; j < 8; ++j) {
    size_t p = pbase + j;
    float v = a * bf2f(upre[p * OUTD + c]) + bb + bf2f(xcur[p * OUTD + c]);
    out[obase + j] = v > 0.f ? v : 0.f;
  }
}

// ---------------------------------------------------------------------------
extern "C" void kernel_launch(void* const* d_in, const int* in_sizes, int n_in,
                              void* d_out, int out_size, void* d_ws, size_t ws_size,
                              hipStream_t stream) {
  const float* lc_xyz   = (const float*)d_in[0];
  const float* lc_feat  = (const float*)d_in[1];
  const float* knn_xyz  = (const float*)d_in[2];
  const float* knn_feat = (const float*)d_in[3];
  const float* w1       = (const float*)d_in[4];
  const float* bn1_g    = (const float*)d_in[5];
  const float* bn1_b    = (const float*)d_in[6];
  const float* wd       = (const float*)d_in[7];
  const float* bd       = (const float*)d_in[8];
  const float* dn_g     = (const float*)d_in[9];
  const float* dn_b     = (const float*)d_in[10];
  const float* wu       = (const float*)d_in[11];
  const float* bu       = (const float*)d_in[12];
  const float* up_g     = (const float*)d_in[13];
  const float* up_b     = (const float*)d_in[14];
  float* out = (float*)d_out;

  // workspace layout
  char* base = (char*)d_ws;
  float* hdr = (float*)base;                                   // 16 KB header
  __bf16* w1bf = (__bf16*)(base + 16 * 1024);                  // 128x160
  __bf16* wdbf = (__bf16*)(base + 16 * 1024 + 40960);          // 2x64x128
  __bf16* wubf = (__bf16*)(base + 16 * 1024 + 40960 + 32768);  // 2x128x64
  float*  gw   = (float*)(base + 131072);                      // NPTS f32
  size_t off = 4u << 20;
  __bf16* PRE  = (__bf16*)(base + off);                        // NPTS x 128 bf16
  off += (size_t)NPTS * OUTD * 2;
  __bf16* HPRE = (__bf16*)(base + off);                        // NPTS x 64 bf16
  off += (size_t)NPTS * MIDD * 2;
  __bf16* XCUR = (__bf16*)(base + off);                        // NPTS x 128 bf16

  // stats stage s: sum[128], sumsq[128] ; affine stage s: a[128], b[128]
  float* ST0 = hdr + 16;
  float* AB0 = hdr + 16 + 5 * 256;
#define ST(s) (ST0 + (s) * 256)
#define AB(s) (AB0 + (s) * 256)

  hipMemsetAsync(d_ws, 0, 16 * 1024, stream);   // zero stat accumulators

  k_relstats<<<1024, 256, 0, stream>>>(knn_xyz, lc_xyz, hdr);
  k_std<<<1, 32, 0, stream>>>(hdr);
  k_gaussw<<<NPTS / 256, 256, 0, stream>>>(knn_xyz, lc_xyz, hdr, gw);
  k_prep<<<(OUTD * CKPAD + 2 * 64 * 128 + 2 * 128 * 64 + 255) / 256, 256, 0, stream>>>(
      w1, wd, wu, w1bf, wdbf, wubf);

  // stage 0: x_pre = w1 * expanded ; bn1 stats
  k_gemm1<<<NPTS / 64, 256, 0, stream>>>(knn_feat, lc_feat, w1bf, PRE, ST(0));
  k_finalize<<<1, 128, 0, stream>>>(ST(0), bn1_g, bn1_b, AB(0), 128);

  // block 0
  k_down<<<NPTS / 64, 256, 0, stream>>>(PRE, AB(0), gw, nullptr, XCUR,
                                        wdbf, bd, HPRE, ST(1));
  k_finalize<<<1, 128, 0, stream>>>(ST(1), dn_g, dn_b, AB(1), 64);
  k_up<<<NPTS / 64, 256, 0, stream>>>(HPRE, AB(1), wubf, bu, PRE, ST(2));
  k_finalize<<<1, 128, 0, stream>>>(ST(2), up_g, up_b, AB(2), 128);

  // block 1 (residual read+write of XCUR is elementwise in-place safe)
  k_down<<<NPTS / 64, 256, 0, stream>>>(PRE, AB(2), nullptr, XCUR, XCUR,
                                        wdbf + 64 * 128, bd + 64, HPRE, ST(3));
  k_finalize<<<1, 128, 0, stream>>>(ST(3), dn_g + 64, dn_b + 64, AB(3), 64);
  k_up<<<NPTS / 64, 256, 0, stream>>>(HPRE, AB(3), wubf + 128 * 64, bu + 128,
                                      PRE, ST(4));
  k_finalize<<<1, 128, 0, stream>>>(ST(4), up_g + 128, up_b + 128, AB(4), 128);

  k_final<<<(BSZ * OUTD * GSZ * (KSZ / 8)) / 256, 256, 0, stream>>>(PRE, XCUR, AB(4), out);
#undef ST
#undef AB
  (void)in_sizes; (void)n_in; (void)out_size; (void)ws_size;
}
